// EegMambaJEPA_69758858821819
// MI455X (gfx1250) — compile-verified, matrix operand order checked by reading, not run on hardware
//
#include <hip/hip_runtime.h>
#include <hip/hip_bf16.h>
#include <math.h>

// ---------------- model dimensions ----------------
#define NLAYER 8
#define BATCH  4
#define CIN    129
#define TIN    5000
#define PATCHW 10
#define LSEQ   501          // 500 patches + cls
#define DMODEL 256
#define DIN    512
#define NSTATE 16
#define DTRANK 32
#define MROWS  (BATCH*LSEQ) // 2004

typedef __attribute__((ext_vector_type(16))) __bf16 v16bf;
typedef __attribute__((ext_vector_type(8)))  __bf16 v8bf;
typedef __attribute__((ext_vector_type(8)))  float  v8f;

__device__ __forceinline__ float silu_f(float x) { return x / (1.f + __expf(-x)); }

// Fragment load for the CDNA5 16-bit 16x32 layout:
// lane<16 -> row r, K {k0..k0+7, k0+16..k0+23}; lane>=16 -> K {+8, +24}.
// p already points at row_base + half*8 + k0.  Two 16B loads, no converts.
__device__ __forceinline__ v16bf load_bf_frag(const __bf16* __restrict__ p) {
  v8bf lo = *(const v8bf*)p;
  v8bf hi = *(const v8bf*)(p + 16);
  v16bf f;
#pragma unroll
  for (int i = 0; i < 8; ++i) { f[i] = lo[i]; f[8 + i] = hi[i]; }
  return f;
}

// ---------------- WMMA GEMM:  C[M,N](f32) = A[M,K](bf16) @ W[N,K](bf16)^T
// (+bias)(softplus)(+resid)(optional bf16 secondary output)
// One wave -> 32x64 tile: 2 A-frags x 4 B-frags = 8 v_wmma per K-step.
// N % 64 == 0, K % 32 == 0 guaranteed. Tail M-tiles over-read inside workspace;
// outputs are masked on store (WMMA rows are independent in M).
__global__ void gemm_wmma_kernel(const __bf16* __restrict__ A, int lda,
                                 const __bf16* __restrict__ W, int ldw,
                                 const float* __restrict__ bias,
                                 const float* __restrict__ resid,
                                 float* __restrict__ C, __bf16* __restrict__ Cbf, int ldc,
                                 int M_, int N_, int K_, int act,
                                 int totalTiles, int tilesM) {
  int wave = blockIdx.x * (blockDim.x >> 5) + (threadIdx.x >> 5);
  if (wave >= totalTiles) return;
  int tm = wave % tilesM, tn = wave / tilesM;
  int m0 = tm * 32, n0 = tn * 64;
  int lane = threadIdx.x & 31;
  int r = lane & 15, half = lane >> 4;

  const __bf16* Ap0 = A + (size_t)(m0 + r) * lda + half * 8;        // rows m0+r
  const __bf16* Ap1 = A + (size_t)(m0 + 16 + r) * lda + half * 8;   // rows m0+16+r
  const __bf16* Wp  = W + (size_t)(n0 + r) * ldw + half * 8;

  v8f acc[8] = {};
  for (int k0 = 0; k0 < K_; k0 += 32) {
    v16bf a0 = load_bf_frag(Ap0 + k0);
    v16bf a1 = load_bf_frag(Ap1 + k0);
    v16bf b0 = load_bf_frag(Wp + k0);
    v16bf b1 = load_bf_frag(Wp + (size_t)16 * ldw + k0);
    v16bf b2 = load_bf_frag(Wp + (size_t)32 * ldw + k0);
    v16bf b3 = load_bf_frag(Wp + (size_t)48 * ldw + k0);

    acc[0] = __builtin_amdgcn_wmma_f32_16x16x32_bf16(false, a0, false, b0, (short)0, acc[0], false, false);
    acc[1] = __builtin_amdgcn_wmma_f32_16x16x32_bf16(false, a0, false, b1, (short)0, acc[1], false, false);
    acc[2] = __builtin_amdgcn_wmma_f32_16x16x32_bf16(false, a0, false, b2, (short)0, acc[2], false, false);
    acc[3] = __builtin_amdgcn_wmma_f32_16x16x32_bf16(false, a0, false, b3, (short)0, acc[3], false, false);
    acc[4] = __builtin_amdgcn_wmma_f32_16x16x32_bf16(false, a1, false, b0, (short)0, acc[4], false, false);
    acc[5] = __builtin_amdgcn_wmma_f32_16x16x32_bf16(false, a1, false, b1, (short)0, acc[5], false, false);
    acc[6] = __builtin_amdgcn_wmma_f32_16x16x32_bf16(false, a1, false, b2, (short)0, acc[6], false, false);
    acc[7] = __builtin_amdgcn_wmma_f32_16x16x32_bf16(false, a1, false, b3, (short)0, acc[7], false, false);
  }

  // ---- store: rows m0 + g*16 + v + half*8, cols n0 + j*16 + r ----
#pragma unroll
  for (int g = 0; g < 2; ++g) {
#pragma unroll
    for (int j = 0; j < 4; ++j) {
      int col = n0 + j * 16 + r;
      float bval = bias ? bias[col] : 0.f;
      v8f cc = acc[g * 4 + j];
#pragma unroll
      for (int v = 0; v < 8; ++v) {
        int row = m0 + g * 16 + v + half * 8;
        if (row < M_) {
          float val = cc[v] + bval;
          if (act == 1) val = (val > 20.f) ? val : log1pf(__expf(val));  // softplus
          if (resid) val += resid[(size_t)row * ldc + col];
          C[(size_t)row * ldc + col] = val;
          if (Cbf) Cbf[(size_t)row * ldc + col] = (__bf16)val;
        }
      }
    }
  }
}

// ---------------- f32 -> bf16 weight conversion (once per launch)
__global__ void cvt_f32_bf16_kernel(const float* __restrict__ src, __bf16* __restrict__ dst, int n) {
  int i = blockIdx.x * 256 + threadIdx.x;
  if (i < n) dst[i] = (__bf16)src[i];
}

// ---------------- patch embedding + cls token -> h (B, LSEQ, 256)
__global__ void patch_embed_kernel(const float* __restrict__ x,
                                   const float* __restrict__ pw,
                                   const float* __restrict__ pb,
                                   const float* __restrict__ cls,
                                   float* __restrict__ h) {
  int idx = blockIdx.x * 256 + threadIdx.x;
  if (idx >= BATCH * LSEQ * DMODEL) return;
  int c = idx & (DMODEL - 1);
  int l = (idx >> 8) % LSEQ;
  int b = idx / (LSEQ * DMODEL);
  if (l == 0) { h[idx] = cls[c]; return; }
  int p = l - 1;
  const float* xb = x + (size_t)b * CIN * TIN + p * PATCHW;
  const float* w  = pw + (size_t)c * CIN * PATCHW;
  float s = pb[c];
  for (int i = 0; i < CIN; ++i) {
    const float* xi = xb + (size_t)i * TIN;
    const float* wi = w + i * PATCHW;
#pragma unroll
    for (int k = 0; k < PATCHW; ++k) s += xi[k] * wi[k];
  }
  h[idx] = s;
}

// ---------------- LayerNorm over last dim (256); writes f32 + bf16 copy
__global__ void layernorm_kernel(const float* __restrict__ x, const float* __restrict__ g,
                                 const float* __restrict__ b, float* __restrict__ y,
                                 __bf16* __restrict__ ybf) {
  __shared__ float s1[256], s2[256];
  int row = blockIdx.x;
  int c = threadIdx.x;
  float v = x[(size_t)row * DMODEL + c];
  s1[c] = v; s2[c] = v * v;
  __syncthreads();
  for (int off = 128; off > 0; off >>= 1) {
    if (c < off) { s1[c] += s1[c + off]; s2[c] += s2[c + off]; }
    __syncthreads();
  }
  float mean = s1[0] * (1.f / DMODEL);
  float var  = s2[0] * (1.f / DMODEL) - mean * mean;
  float out = (v - mean) * rsqrtf(var + 1e-5f) * g[c] + b[c];
  y[(size_t)row * DMODEL + c] = out;
  ybf[(size_t)row * DMODEL + c] = (__bf16)out;
}

// final LN applied only to the cls row (l==0) of each batch -> d_out
__global__ void final_ln_kernel(const float* __restrict__ h, const float* __restrict__ g,
                                const float* __restrict__ b, float* __restrict__ out) {
  __shared__ float s1[256], s2[256];
  int row = blockIdx.x * LSEQ;   // l = 0
  int c = threadIdx.x;
  float v = h[(size_t)row * DMODEL + c];
  s1[c] = v; s2[c] = v * v;
  __syncthreads();
  for (int off = 128; off > 0; off >>= 1) {
    if (c < off) { s1[c] += s1[c + off]; s2[c] += s2[c + off]; }
    __syncthreads();
  }
  float mean = s1[0] * (1.f / DMODEL);
  float var  = s2[0] * (1.f / DMODEL) - mean * mean;
  out[blockIdx.x * DMODEL + c] = (v - mean) * rsqrtf(var + 1e-5f) * g[c] + b[c];
}

// ---------------- causal depthwise conv (K=4, left pad 3) + SiLU; f32 + bf16 outputs
__global__ void dwconv_silu_kernel(const float* __restrict__ x, int ldx,
                                   const float* __restrict__ w, const float* __restrict__ bias,
                                   float* __restrict__ y, __bf16* __restrict__ ybf) {
  int idx = blockIdx.x * 256 + threadIdx.x;
  if (idx >= BATCH * LSEQ * DIN) return;
  int d = idx & (DIN - 1);
  int l = (idx / DIN) % LSEQ;
  int b = idx / (DIN * LSEQ);
  float s = bias[d];
#pragma unroll
  for (int k = 0; k < 4; ++k) {
    int ll = l + k - 3;
    if (ll >= 0) s += x[((size_t)(b * LSEQ + ll)) * ldx + d] * w[d * 4 + k];
  }
  float out = silu_f(s);
  size_t o = ((size_t)(b * LSEQ + l)) * DIN + d;
  y[o] = out;
  ybf[o] = (__bf16)out;
}

// ---------------- flip along L; writes f32 + bf16
__global__ void flip_kernel(const float* __restrict__ x, float* __restrict__ y,
                            __bf16* __restrict__ ybf) {
  int idx = blockIdx.x * 256 + threadIdx.x;
  if (idx >= BATCH * LSEQ * DIN) return;
  int d = idx & (DIN - 1);
  int l = (idx / DIN) % LSEQ;
  int b = idx / (DIN * LSEQ);
  float v = x[((size_t)(b * LSEQ + (LSEQ - 1 - l))) * DIN + d];
  size_t o = ((size_t)(b * LSEQ + l)) * DIN + d;
  y[o] = v;
  ybf[o] = (__bf16)v;
}

// ---------------- selective scan: one thread per (b,d), N=16 state in registers
__global__ void selective_scan_kernel(const float* __restrict__ u, const float* __restrict__ delta,
                                      const float* __restrict__ Al, const float* __restrict__ dtbc,
                                      const float* __restrict__ Dp, float* __restrict__ ys) {
  int t = blockIdx.x * blockDim.x + threadIdx.x;
  if (t >= BATCH * DIN) return;
  int d = t & (DIN - 1);
  int b = t / DIN;
  float a[NSTATE], hst[NSTATE];
#pragma unroll
  for (int n = 0; n < NSTATE; ++n) { a[n] = -__expf(Al[d * NSTATE + n]); hst[n] = 0.f; }
  float Dd = Dp[d];
  for (int l = 0; l < LSEQ; ++l) {
    size_t rb = (size_t)(b * LSEQ + l);
    float dt = delta[rb * DIN + d];
    float ut = u[rb * DIN + d];
    float du = dt * ut;
    const float* bc = dtbc + rb * 64;
    float y = 0.f;
#pragma unroll
    for (int n = 0; n < NSTATE; ++n) {
      float dA = __expf(dt * a[n]);
      hst[n] = dA * hst[n] + du * bc[DTRANK + n];        // Bm at cols 32..47
      y += hst[n] * bc[DTRANK + NSTATE + n];             // C  at cols 48..63
    }
    ys[rb * DIN + d] = y + ut * Dd;
  }
}

// ---------------- out(bf16) = a * silu(z)   (z strided, ldz); GEMM-only consumer
__global__ void mul_silu_kernel(const float* __restrict__ a, const float* __restrict__ z,
                                int ldz, __bf16* __restrict__ out) {
  int idx = blockIdx.x * 256 + threadIdx.x;
  if (idx >= MROWS * DIN) return;
  int d = idx & (DIN - 1);
  int row = idx / DIN;
  float zz = z[(size_t)row * ldz + d];
  out[idx] = (__bf16)(a[idx] * silu_f(zz));
}

// ---------------- ycat(bf16) = [yf * silu(res), yb * silu(res)]  (res strided 1024)
__global__ void gate_concat_kernel(const float* __restrict__ yf, const float* __restrict__ yb,
                                   const float* __restrict__ res, __bf16* __restrict__ ycat) {
  int idx = blockIdx.x * 256 + threadIdx.x;
  if (idx >= MROWS * DIN) return;
  int d = idx & (DIN - 1);
  int row = idx / DIN;
  float rr = res[(size_t)row * 1024 + d];
  float g = silu_f(rr);
  ycat[(size_t)row * 1024 + d] = (__bf16)(yf[idx] * g);
  ycat[(size_t)row * 1024 + DIN + d] = (__bf16)(yb[idx] * g);
}

// ---------------- host-side helpers ----------------
static inline void launch_gemm(const __bf16* A, int lda, const __bf16* W, int ldw,
                               const float* bias, const float* resid,
                               float* C, __bf16* Cbf, int ldc, int M_, int N_, int K_, int act,
                               hipStream_t stream) {
  int tilesM = (M_ + 31) / 32;
  int tilesN = N_ / 64;
  int total = tilesM * tilesN;
  int blocks = (total + 7) / 8;  // 8 waves (256 threads) per block
  gemm_wmma_kernel<<<blocks, 256, 0, stream>>>(A, lda, W, ldw, bias, resid, C, Cbf, ldc,
                                               M_, N_, K_, act, total, tilesM);
}

static inline void launch_cvt(const float* src, __bf16* dst, int n, hipStream_t stream) {
  cvt_f32_bf16_kernel<<<(n + 255) / 256, 256, 0, stream>>>(src, dst, n);
}

extern "C" void kernel_launch(void* const* d_in, const int* in_sizes, int n_in,
                              void* d_out, int out_size, void* d_ws, size_t ws_size,
                              hipStream_t stream) {
  (void)in_sizes; (void)n_in; (void)out_size; (void)ws_size;

  const float* x       = (const float*)d_in[0];
  const float* patch_w = (const float*)d_in[1];
  const float* patch_b = (const float*)d_in[2];
  const float* cls_tok = (const float*)d_in[3];
  const float* ln_g    = (const float*)d_in[4];
  const float* ln_b    = (const float*)d_in[5];
  const float* in_w    = (const float*)d_in[6];
  const float* cw      = (const float*)d_in[7];
  const float* cb      = (const float*)d_in[8];
  const float* out_w   = (const float*)d_in[9];
  const float* fn_g    = (const float*)d_in[10];
  const float* fn_b    = (const float*)d_in[11];
  const float* m_in[2]  = {(const float*)d_in[12], (const float*)d_in[21]};
  const float* m_cw[2]  = {(const float*)d_in[13], (const float*)d_in[22]};
  const float* m_cb[2]  = {(const float*)d_in[14], (const float*)d_in[23]};
  const float* m_xp[2]  = {(const float*)d_in[15], (const float*)d_in[24]};
  const float* m_dtw[2] = {(const float*)d_in[16], (const float*)d_in[25]};
  const float* m_dtb[2] = {(const float*)d_in[17], (const float*)d_in[26]};
  const float* m_Al[2]  = {(const float*)d_in[18], (const float*)d_in[27]};
  const float* m_D[2]   = {(const float*)d_in[19], (const float*)d_in[28]};
  const float* m_ow[2]  = {(const float*)d_in[20], (const float*)d_in[29]};

  // --------- f32 workspace ----------
  float* ws = (float*)d_ws;
  float* h     = ws; ws += (size_t)MROWS * DMODEL;
  float* xn    = ws; ws += (size_t)MROWS * DMODEL;
  float* xcres = ws; ws += (size_t)MROWS * 1024;
  float* xcv   = ws; ws += (size_t)MROWS * DIN;
  float* xflip = ws; ws += (size_t)MROWS * DIN;
  float* xz    = ws; ws += (size_t)MROWS * 1024;
  float* xmc   = ws; ws += (size_t)MROWS * DIN;
  float* dtbc  = ws; ws += (size_t)MROWS * 64;
  float* delta = ws; ws += (size_t)MROWS * DIN;
  float* ysb   = ws; ws += (size_t)MROWS * DIN;
  float* yf    = ws; ws += (size_t)MROWS * DIN;
  float* yb    = ws; ws += (size_t)MROWS * DIN;

  // --------- bf16 weight workspace (converted once per launch) ----------
  const int n_inw  = NLAYER * 1024 * 256;
  const int n_outw = NLAYER * 256 * 1024;
  const int n_min  = NLAYER * 1024 * 512;
  const int n_mxp  = NLAYER * 64 * 512;
  const int n_mdtw = NLAYER * 512 * 32;
  const int n_mow  = NLAYER * 512 * 512;

  __bf16* bw = (__bf16*)ws;
  __bf16* b_inw  = bw; bw += n_inw;
  __bf16* b_outw = bw; bw += n_outw;
  __bf16* b_min[2];  b_min[0]  = bw; bw += n_min;  b_min[1]  = bw; bw += n_min;
  __bf16* b_mxp[2];  b_mxp[0]  = bw; bw += n_mxp;  b_mxp[1]  = bw; bw += n_mxp;
  __bf16* b_mdtw[2]; b_mdtw[0] = bw; bw += n_mdtw; b_mdtw[1] = bw; bw += n_mdtw;
  __bf16* b_mow[2];  b_mow[0]  = bw; bw += n_mow;  b_mow[1]  = bw; bw += n_mow;

  // --------- bf16 activation workspace (GEMM A operands) ----------
  __bf16* xn_bf    = bw; bw += (size_t)MROWS * DMODEL;
  __bf16* xcv_bf   = bw; bw += (size_t)MROWS * DIN;
  __bf16* xflip_bf = bw; bw += (size_t)MROWS * DIN;   // also reused as dummy for yb flip
  __bf16* xmc_bf   = bw; bw += (size_t)MROWS * DIN;
  __bf16* dtbc_bf  = bw; bw += (size_t)MROWS * 64;
  __bf16* ymul_bf  = bw; bw += (size_t)MROWS * DIN;
  __bf16* ycat_bf  = bw; bw += (size_t)MROWS * 1024;
  bw += 65536;  // tail pad: tolerate tail-tile over-reads of last A operand

  launch_cvt(in_w,  b_inw,  n_inw,  stream);
  launch_cvt(out_w, b_outw, n_outw, stream);
  for (int dir = 0; dir < 2; ++dir) {
    launch_cvt(m_in[dir],  b_min[dir],  n_min,  stream);
    launch_cvt(m_xp[dir],  b_mxp[dir],  n_mxp,  stream);
    launch_cvt(m_dtw[dir], b_mdtw[dir], n_mdtw, stream);
    launch_cvt(m_ow[dir],  b_mow[dir],  n_mow,  stream);
  }

  const int ewBlocks = (BATCH * LSEQ * DIN + 255) / 256;

  // 1) patch embed + cls
  patch_embed_kernel<<<(BATCH * LSEQ * DMODEL + 255) / 256, 256, 0, stream>>>(
      x, patch_w, patch_b, cls_tok, h);

  // 2) 8 bidirectional Mamba blocks
  for (int i = 0; i < NLAYER; ++i) {
    layernorm_kernel<<<MROWS, 256, 0, stream>>>(h, ln_g + i * DMODEL, ln_b + i * DMODEL, xn, xn_bf);
    // xn @ in_w^T -> (xc | res)
    launch_gemm(xn_bf, DMODEL, b_inw + (size_t)i * 1024 * 256, 256, nullptr, nullptr,
                xcres, nullptr, 1024, MROWS, 1024, 256, 0, stream);
    // xc = silu(dwconv(xc))
    dwconv_silu_kernel<<<ewBlocks, 256, 0, stream>>>(xcres, 1024, cw + i * DIN * 4, cb + i * DIN,
                                                     xcv, xcv_bf);

    for (int dir = 0; dir < 2; ++dir) {
      const __bf16* u_bf = xcv_bf;
      if (dir == 1) {
        flip_kernel<<<ewBlocks, 256, 0, stream>>>(xcv, xflip, xflip_bf);
        u_bf = xflip_bf;
      }
      // u @ m_in^T -> (xm | z)
      launch_gemm(u_bf, DIN, b_min[dir] + (size_t)i * 1024 * 512, 512, nullptr, nullptr,
                  xz, nullptr, 1024, MROWS, 1024, 512, 0, stream);
      // xm = silu(dwconv(xm))
      dwconv_silu_kernel<<<ewBlocks, 256, 0, stream>>>(xz, 1024, m_cw[dir] + i * DIN * 4,
                                                       m_cb[dir] + i * DIN, xmc, xmc_bf);
      // xm @ xp^T -> (dt | Bm | C), f32 for the scan + bf16 for the delta GEMM
      launch_gemm(xmc_bf, DIN, b_mxp[dir] + (size_t)i * 64 * 512, 512, nullptr, nullptr,
                  dtbc, dtbc_bf, 64, MROWS, 64, 512, 0, stream);
      // delta = softplus(dt @ dtw^T + dtb)
      launch_gemm(dtbc_bf, 64, b_mdtw[dir] + (size_t)i * 512 * 32, 32, m_dtb[dir] + i * DIN, nullptr,
                  delta, nullptr, DIN, MROWS, DIN, 32, 1, stream);
      // selective scan
      selective_scan_kernel<<<(BATCH * DIN + 255) / 256, 256, 0, stream>>>(
          xmc, delta, m_Al[dir] + (size_t)i * DIN * NSTATE, dtbc, m_D[dir] + i * DIN, ysb);
      // ys * silu(z) -> bf16
      mul_silu_kernel<<<ewBlocks, 256, 0, stream>>>(ysb, xz + DIN, 1024, ymul_bf);
      // @ ow^T
      if (dir == 0) {
        launch_gemm(ymul_bf, DIN, b_mow[dir] + (size_t)i * 512 * 512, 512, nullptr, nullptr,
                    yf, nullptr, DIN, MROWS, DIN, 512, 0, stream);
      } else {
        launch_gemm(ymul_bf, DIN, b_mow[dir] + (size_t)i * 512 * 512, 512, nullptr, nullptr,
                    ysb, nullptr, DIN, MROWS, DIN, 512, 0, stream);
        flip_kernel<<<ewBlocks, 256, 0, stream>>>(ysb, yb, xflip_bf /* dummy bf16 sink */);
      }
    }
    // gate + concat -> bf16 ycat
    gate_concat_kernel<<<ewBlocks, 256, 0, stream>>>(yf, yb, xcres + DIN, ycat_bf);
    // h = ycat @ out_w^T + h  (residual fused)
    launch_gemm(ycat_bf, 1024, b_outw + (size_t)i * 256 * 1024, 1024, nullptr, h,
                h, nullptr, DMODEL, MROWS, DMODEL, 1024, 0, stream);
  }

  // 3) final LN on cls rows -> output (4,256)
  final_ln_kernel<<<BATCH, 256, 0, stream>>>(h, fn_g, fn_b, (float*)d_out);
}
